// NeuralODE_35321811042623
// MI455X (gfx1250) — matrix operand order repaired and checked by reading, not compiled, not run on hardware
//
#include <hip/hip_runtime.h>
#include <math.h>

// ---------------------------------------------------------------------------
// NeuralODE for MI455X (gfx1250): bf16 WMMA GEMMs, fp32 accumulate/epilogue.
// ---------------------------------------------------------------------------

typedef unsigned short ushort_t;
typedef __attribute__((ext_vector_type(16))) __bf16 v16bf;
typedef __attribute__((ext_vector_type(8)))  float  v8f;

#define BS     128
#define HWC    3136      // H*W
#define AREG   9408      // 3*H*W
#define LSEQ   64
#define DEMB   128
#define ZDIM   17728     // 3*H*W + L*D + D
#define INDIM  17740     // ZDIM + 12
#define K1C    555       // ceil(INDIM/32) -> K padded to 17760
#define HID    4096
#define K2C    128       // 4096/32
#define NT1    256       // HID/16
#define NT2    1108      // ZDIM/16
#define NROW   12
#define NSTEP  11

// round-to-nearest-even fp32 -> bf16 (stored as raw u16)
__device__ __forceinline__ ushort_t f2bf(float f) {
  unsigned u = __float_as_uint(f);
  u += 0x7fffu + ((u >> 16) & 1u);
  return (ushort_t)(u >> 16);
}

// ---------------------------------------------------------------------------
// Weight packing: fp32 row-major -> bf16 WMMA B-fragment blocks.
// Block layout (1KB): blk = nt*KC + kc ; elem = lane*16 + j, where
// B-frag (32x16, 16-bit): lanes 0-15 hold K=0..15 (j=k&15, n=lane),
//                         lanes 16-31 hold K=16..31.
// ---------------------------------------------------------------------------
__global__ __launch_bounds__(256) void pack_w1_kernel(
    const float* __restrict__ W1, ushort_t* __restrict__ Wp) {
  long long i = (long long)blockIdx.x * 256 + threadIdx.x;  // < 555*256*512
  int t = (int)(i & 511);
  long long blk = i >> 9;                 // nt*K1C + kc
  int kc = (int)(blk % K1C);
  int nt = (int)(blk / K1C);
  int n_off = t & 15;
  int k_off = t >> 4;                     // 0..31
  int k = kc * 32 + k_off;
  int n = nt * 16 + n_off;
  float v = (k < INDIM) ? W1[(long long)k * HID + n] : 0.0f;
  int lane = n_off + (k_off & 16);
  int j = k_off & 15;
  Wp[(blk << 9) + lane * 16 + j] = f2bf(v);
}

__global__ __launch_bounds__(256) void pack_w2_kernel(
    const float* __restrict__ W2, ushort_t* __restrict__ Wp) {
  long long i = (long long)blockIdx.x * 256 + threadIdx.x;  // < 128*1108*512
  int t = (int)(i & 511);
  long long blk = i >> 9;                 // nt*K2C + kc
  int kc = (int)(blk & (K2C - 1));
  int nt = (int)(blk >> 7);
  int n_off = t & 15;
  int k_off = t >> 4;
  int k = kc * 32 + k_off;                // always < 4096
  int n = nt * 16 + n_off;                // always < 17728
  float v = W2[(long long)k * ZDIM + n];
  int lane = n_off + (k_off & 16);
  int j = k_off & 15;
  Wp[(blk << 9) + lane * 16 + j] = f2bf(v);
}

// ---------------------------------------------------------------------------
// z0 assembly: [tiled freq | gathered+transposed seq embeddings | user emb]
// ---------------------------------------------------------------------------
__global__ __launch_bounds__(256) void build_z0_kernel(
    const float* __restrict__ freq, const int* __restrict__ seq,
    const int* __restrict__ uid, const float* __restrict__ femb,
    const int* __restrict__ n_poi_p, float* __restrict__ zbuf) {
  int i = blockIdx.x * 256 + threadIdx.x;  // < 128*17728
  int bs = i / ZDIM;
  int c = i - bs * ZDIM;
  float v;
  if (c < AREG) {
    int pos = c % HWC;
    v = freq[bs * HWC + pos];
  } else if (c < AREG + LSEQ * DEMB) {
    int o = c - AREG;
    int d = o >> 6;        // / LSEQ
    int l = o & 63;        // % LSEQ
    int poi = seq[bs * LSEQ + l];
    v = femb[(long long)poi * DEMB + d];
  } else {
    int d = c - (AREG + LSEQ * DEMB);
    v = femb[(long long)(n_poi_p[0] + uid[bs]) * DEMB + d];
  }
  zbuf[i] = v;
}

// ---------------------------------------------------------------------------
// Time grid + sin/cos embedding (12 x BS x 6)
// ---------------------------------------------------------------------------
__global__ void time_emb_kernel(const float* __restrict__ cur,
                                const float* __restrict__ tar,
                                float* __restrict__ emb) {
  int i = blockIdx.x * 256 + threadIdx.x;
  if (i >= NROW * BS) return;
  int r = i / BS, bs = i - r * BS;
  const float P[3] = {24.0f, 7.0f, 365.0f};
  int jf = r >> 2, kk = r & 3;
  float fr = (float)kk * (1.0f / 3.0f);
  #pragma unroll
  for (int f = 0; f < 3; ++f) {
    float val;
    if (f < jf) {
      val = tar[bs * 3 + f];
    } else if (f > jf) {
      val = cur[bs * 3 + f];
    } else {
      float s = cur[bs * 3 + f], e = tar[bs * 3 + f];
      bool wrap = s > e;
      float ea = wrap ? e + P[f] : e;
      float inter = s + (ea - s) * fr;
      val = wrap ? fmodf(inter, P[f]) : inter;
    }
    float ph = 6.283185307179586f * val / P[f];
    emb[(r * BS + bs) * 6 + f]     = sinf(ph);
    emb[(r * BS + bs) * 6 + 3 + f] = cosf(ph);
  }
}

// ---------------------------------------------------------------------------
// A-fragment packing (16x32, 16-bit, k order matching B packing):
//  lanes 0-15 : M=lane,    elems 0-15 = K 0..15
//  lanes16-31 : M=lane-16, elems 0-15 = K 16..31
// inverse: lane = m + (k&16) ; j = k&15
// ---------------------------------------------------------------------------
__global__ __launch_bounds__(256) void pack_a1_kernel(
    const float* __restrict__ zbuf, const float* __restrict__ emb0,
    const float* __restrict__ emb1, ushort_t* __restrict__ Ap) {
  int i = blockIdx.x * 256 + threadIdx.x;  // < 8*555*512
  int t = i & 511;
  int blk = i >> 9;            // mt*K1C + kc
  int kc = blk % K1C;
  int mt = blk / K1C;
  int m_off = t >> 5;
  int k_off = t & 31;
  int m = mt * 16 + m_off;
  int k = kc * 32 + k_off;
  float v;
  if (k < ZDIM)            v = zbuf[(long long)m * ZDIM + k];
  else if (k < ZDIM + 6)   v = emb0[m * 6 + (k - ZDIM)];
  else if (k < ZDIM + 12)  v = emb1[m * 6 + (k - ZDIM - 6)];
  else                     v = 0.0f;
  int lane = m_off + (k_off & 16);
  int j = k_off & 15;
  Ap[((long long)blk << 9) + lane * 16 + j] = f2bf(v);
}

__global__ __launch_bounds__(256) void pack_a2_kernel(
    const float* __restrict__ hbuf, ushort_t* __restrict__ Ap) {
  int i = blockIdx.x * 256 + threadIdx.x;  // < 8*128*512
  int t = i & 511;
  int blk = i >> 9;            // mt*K2C + kc
  int kc = blk & (K2C - 1);
  int mt = blk >> 7;
  int m_off = t >> 5;
  int k_off = t & 31;
  int m = mt * 16 + m_off;
  int k = kc * 32 + k_off;
  float v = hbuf[m * HID + k];
  int lane = m_off + (k_off & 16);
  int j = k_off & 15;
  Ap[((long long)blk << 9) + lane * 16 + j] = f2bf(v);
}

// ---------------------------------------------------------------------------
// GEMM1: h = tanh(inp @ W1 + b1)   (128 x 17760pad) * (17760 x 4096)
// block = 8 waves (one m-tile each), blockIdx.x = n-tile; all waves share W1p.
// ---------------------------------------------------------------------------
__global__ __launch_bounds__(256) void gemm1_kernel(
    const ushort_t* __restrict__ Ap, const ushort_t* __restrict__ Wp,
    const float* __restrict__ b1, float* __restrict__ h) {
  int lane = threadIdx.x & 31;
  int mt = threadIdx.x >> 5;     // 0..7
  int nt = blockIdx.x;           // 0..255
  const v16bf* a = (const v16bf*)(Ap + (((long long)mt * K1C) << 9) + lane * 16);
  const v16bf* b = (const v16bf*)(Wp + (((long long)nt * K1C) << 9) + lane * 16);
  v8f acc = {0.f, 0.f, 0.f, 0.f, 0.f, 0.f, 0.f, 0.f};
  #pragma unroll 4
  for (int kc = 0; kc < K1C; ++kc) {
    v16bf av = a[kc * 32];       // 32 v16bf per 1KB block
    v16bf bv = b[kc * 32];
    acc = __builtin_amdgcn_wmma_f32_16x16x32_bf16(false, av, false, bv,
                                                  (short)0, acc, false, false);
  }
  // C layout: lanes 0-15 N=lane rows M=r; lanes 16-31 N=lane-16 rows M=8+r
  int n = nt * 16 + (lane & 15);
  float bias = b1[n];
  int m_base = mt * 16 + ((lane >> 4) << 3);
  #pragma unroll
  for (int r = 0; r < 8; ++r)
    h[(m_base + r) * HID + n] = tanhf(acc[r] + bias);
}

// ---------------------------------------------------------------------------
// GEMM2: z += h @ W2 + b2   (128 x 4096) * (4096 x 17728), in-place z update
// ---------------------------------------------------------------------------
__global__ __launch_bounds__(256) void gemm2_kernel(
    const ushort_t* __restrict__ Ap, const ushort_t* __restrict__ Wp,
    const float* __restrict__ b2, float* __restrict__ zbuf) {
  int lane = threadIdx.x & 31;
  int mt = threadIdx.x >> 5;     // 0..7
  int nt = blockIdx.x;           // 0..1107
  const v16bf* a = (const v16bf*)(Ap + (((long long)mt * K2C) << 9) + lane * 16);
  const v16bf* b = (const v16bf*)(Wp + (((long long)nt * K2C) << 9) + lane * 16);
  v8f acc = {0.f, 0.f, 0.f, 0.f, 0.f, 0.f, 0.f, 0.f};
  #pragma unroll 4
  for (int kc = 0; kc < K2C; ++kc) {
    acc = __builtin_amdgcn_wmma_f32_16x16x32_bf16(false, a[kc * 32], false,
                                                  b[kc * 32], (short)0, acc,
                                                  false, false);
  }
  int n = nt * 16 + (lane & 15);
  float bias = b2[n];
  int m_base = mt * 16 + ((lane >> 4) << 3);
  #pragma unroll
  for (int r = 0; r < 8; ++r) {
    long long idx = (long long)(m_base + r) * ZDIM + n;
    zbuf[idx] = zbuf[idx] + acc[r] + bias;
  }
}

__global__ __launch_bounds__(256) void copy_out_kernel(
    const float* __restrict__ zbuf, float* __restrict__ out) {
  int i = blockIdx.x * 256 + threadIdx.x;  // exactly 128*17728 threads
  out[i] = zbuf[i];
}

// ---------------------------------------------------------------------------
extern "C" void kernel_launch(void* const* d_in, const int* in_sizes, int n_in,
                              void* d_out, int out_size, void* d_ws, size_t ws_size,
                              hipStream_t stream) {
  const float* input_freq = (const float*)d_in[0];
  const int*   input_seq  = (const int*)d_in[1];
  const int*   uid        = (const int*)d_in[2];
  const float* cur_time   = (const float*)d_in[3];
  const float* tar_time   = (const float*)d_in[4];
  const float* fuse_embed = (const float*)d_in[5];
  const float* W1         = (const float*)d_in[6];
  const float* b1         = (const float*)d_in[7];
  const float* W2         = (const float*)d_in[8];
  const float* b2         = (const float*)d_in[9];
  const int*   n_poi      = (const int*)d_in[10];
  float* out = (float*)d_out;

  // workspace carve-up (256B aligned)
  char* base = (char*)d_ws;
  size_t off = 0;
  #define WS_ALLOC(ptrvar, type, bytes)                         \
    type* ptrvar = (type*)(base + off);                         \
    off = (off + (size_t)(bytes) + 255) & ~(size_t)255;
  WS_ALLOC(W1p, ushort_t, (size_t)K1C * NT1 * 512 * 2)   // 145.5 MB
  WS_ALLOC(W2p, ushort_t, (size_t)K2C * NT2 * 512 * 2)   // 145.2 MB
  WS_ALLOC(A1p, ushort_t, (size_t)8 * K1C * 512 * 2)     // 4.5 MB
  WS_ALLOC(A2p, ushort_t, (size_t)8 * K2C * 512 * 2)     // 1 MB
  WS_ALLOC(zbuf, float,   (size_t)BS * ZDIM * 4)         // 9.1 MB
  WS_ALLOC(hbuf, float,   (size_t)BS * HID * 4)          // 2 MB
  WS_ALLOC(embbuf, float, (size_t)NROW * BS * 6 * 4)
  #undef WS_ALLOC

  // one-shot per call: convert + swizzle weights to bf16 fragment blocks
  pack_w1_kernel<<<K1C * NT1 * 2, 256, 0, stream>>>(W1, W1p);
  pack_w2_kernel<<<K2C * NT2 * 2, 256, 0, stream>>>(W2, W2p);

  build_z0_kernel<<<(BS * ZDIM) / 256, 256, 0, stream>>>(
      input_freq, input_seq, uid, fuse_embed, n_poi, zbuf);
  time_emb_kernel<<<(NROW * BS + 255) / 256, 256, 0, stream>>>(
      cur_time, tar_time, embbuf);

  for (int step = 0; step < NSTEP; ++step) {
    const float* e0 = embbuf + (size_t)step * BS * 6;
    const float* e1 = embbuf + (size_t)(step + 1) * BS * 6;
    pack_a1_kernel<<<8 * K1C * 2, 256, 0, stream>>>(zbuf, e0, e1, A1p);
    gemm1_kernel<<<NT1, 256, 0, stream>>>(A1p, W1p, b1, hbuf);
    pack_a2_kernel<<<8 * K2C * 2, 256, 0, stream>>>(hbuf, A2p);
    gemm2_kernel<<<NT2, 256, 0, stream>>>(A2p, W2p, b2, zbuf);
  }

  copy_out_kernel<<<(BS * ZDIM) / 256, 256, 0, stream>>>(zbuf, out);
}
